// EnsembleWorldModel_9002251452628
// MI455X (gfx1250) — matrix operand order repaired and checked by reading, not compiled, not run on hardware
//
#include <hip/hip_runtime.h>
#include <hip/hip_bf16.h>

// ---------------------------------------------------------------------------
// EnsembleWorldModel forward for MI455X (gfx1250, wave32, WMMA).
// bf16 WMMA GEMMs (v_wmma_f32_16x16x32_bf16), f32 accumulate, f32 LN/GELU.
// Memory-bound (~2-3 GB HBM traffic @ 23.3 TB/s >> WMMA math time):
//   - weights transposed+converted f32->bf16 once per call (halves L2 re-read
//     bytes, no conversion VALU in the GEMM hot loop),
//   - GEMM LDS tiles double-buffered; tiles staged with CDNA5
//     GLOBAL_LOAD_ASYNC_TO_LDS_B128 (ASYNCcnt) -> zero staging VGPRs (fixes
//     the scratch spill seen with register staging), with a register-staged
//     fallback under a loosened register budget.
// ---------------------------------------------------------------------------

typedef __attribute__((ext_vector_type(16))) __bf16 v16bf;
typedef __attribute__((ext_vector_type(8)))  float  v8f;
typedef __attribute__((ext_vector_type(4)))  int    ewm_v4i;

union V16BF { uint4 q[2]; v16bf v; };
static_assert(sizeof(v16bf) == 32, "v16bf must be 32 bytes");

__device__ __forceinline__ unsigned short f2bf_rne(float f) {
    unsigned u = __float_as_uint(f);
    unsigned r = u + 0x7FFFu + ((u >> 16) & 1u);
    return (unsigned short)(r >> 16);
}

#if __has_builtin(__builtin_amdgcn_global_load_async_to_lds_b128)
#define EWM_HAVE_ASYNC_LDS 1
__device__ __forceinline__ void ewm_async_cp16(const void* gsrc, void* ldst) {
    __builtin_amdgcn_global_load_async_to_lds_b128(
        (ewm_v4i __attribute__((address_space(1)))*)gsrc,
        (ewm_v4i __attribute__((address_space(3)))*)ldst, 0, 0);
}
__device__ __forceinline__ void ewm_wait_async() {
#if __has_builtin(__builtin_amdgcn_s_wait_asynccnt)
    __builtin_amdgcn_s_wait_asynccnt(0);
#else
    asm volatile("s_wait_asynccnt 0" ::: "memory");
#endif
}
#else
#define EWM_HAVE_ASYNC_LDS 0
#endif

// ---------------------------------------------------------------------------
// Pack x = concat(state, action) as bf16 [B, 576]
// ---------------------------------------------------------------------------
__global__ __launch_bounds__(256)
void ewm_pack_x(const float* __restrict__ state, const float* __restrict__ action,
                unsigned short* __restrict__ xcat, int Bv, int Sv, int Av) {
    int idx = blockIdx.x * 256 + threadIdx.x;
    int SA = Sv + Av;
    if (idx >= Bv * SA) return;
    int b = idx / SA, c = idx - b * SA;
    float f = (c < Sv) ? state[(size_t)b * Sv + c] : action[(size_t)b * Av + (c - Sv)];
    xcat[idx] = f2bf_rne(f);
}

// ---------------------------------------------------------------------------
// Transpose + convert: W (f32, [K][N] row-major) -> Wt (bf16, [N][K] row-major)
// ---------------------------------------------------------------------------
__global__ __launch_bounds__(256)
void ewm_wtrans_bf16(const float* __restrict__ W, unsigned short* __restrict__ Wt,
                     int Kdim, int Ndim) {
    __shared__ unsigned short tile[32][33];
    const float* Wm = W + (size_t)blockIdx.z * Kdim * Ndim;
    unsigned short* Wtm = Wt + (size_t)blockIdx.z * Kdim * Ndim;
    const int n0 = blockIdx.x * 32, k0 = blockIdx.y * 32;
    const int tx = threadIdx.x & 31, ty = threadIdx.x >> 5;
#pragma unroll
    for (int r = ty; r < 32; r += 8)
        tile[r][tx] = f2bf_rne(Wm[(size_t)(k0 + r) * Ndim + (n0 + tx)]);
    __syncthreads();
#pragma unroll
    for (int r = ty; r < 32; r += 8)
        Wtm[(size_t)(n0 + r) * Kdim + (k0 + tx)] = tile[tx][r];
}

// ---------------------------------------------------------------------------
// Ensemble GEMM: C[e] = A[e] (bf16, MxK row-major) * Wt[e]^T (bf16, [N][K]) + bias[e]
// Block tile 128x128, 4 waves x (64x64), 16x16x32 bf16 WMMA, double-buffered LDS.
// ---------------------------------------------------------------------------
__global__ __launch_bounds__(128, 1)
void ewm_gemm_bias(const unsigned short* __restrict__ A, long long aStrideE,
                   const unsigned short* __restrict__ Wt, long long wStrideE,
                   const float* __restrict__ bias, long long biasStrideE,
                   float* __restrict__ C, long long cStrideE,
                   int M, int N, int K) {
    __shared__ unsigned short lds_a[2][128 * 40];   // A tile [128 m][32 k], pitch 40
    __shared__ unsigned short lds_b[2][128 * 40];   // Wt tile [128 n][32 k], pitch 40

    const int e    = blockIdx.z;
    const int tid  = threadIdx.x;
    const int wave = tid >> 5;
    const int lane = tid & 31;
    const int m0   = blockIdx.y * 128;
    const int n0   = blockIdx.x * 128;

    const unsigned short* Ae  = A  + (size_t)e * aStrideE;
    const unsigned short* Wte = Wt + (size_t)e * wStrideE;
    const float* biase = bias + (size_t)e * biasStrideE;
    float*       Ce    = C    + (size_t)e * cStrideE;

    const int wrow = (wave >> 1) * 64;
    const int wcol = (wave & 1) * 64;

    v8f acc[4][4];
#pragma unroll
    for (int mi = 0; mi < 4; ++mi)
#pragma unroll
        for (int ni = 0; ni < 4; ++ni)
#pragma unroll
            for (int r = 0; r < 8; ++r) acc[mi][ni][r] = 0.0f;

    const int hl    = lane & 15;
    const int kgrpA = (lane >> 4) * 8;    // A frag: lanes16-31 K base +8
    const int kgrpB = (lane >> 4) * 8;    // B frag: lanes16-31 K base +8

    const char* gAp = (const char*)(Ae  + (size_t)(m0 + tid) * K);
    const char* gBp = (const char*)(Wte + (size_t)(n0 + tid) * K);

#if EWM_HAVE_ASYNC_LDS
    // -------- async global->LDS staging: no VGPR staging, ASYNCcnt-tracked --
    char* la = (char*)&lds_a[0][tid * 40];
    char* lb = (char*)&lds_b[0][tid * 40];
    const int LBUF = 128 * 40 * 2;   // bytes between buffer 0 and buffer 1

    // issue tile 0 into buffer 0 (each lane copies its own 64B row slab)
#pragma unroll
    for (int i = 0; i < 4; ++i) {
        ewm_async_cp16(gAp + i * 16, la + i * 16);
        ewm_async_cp16(gBp + i * 16, lb + i * 16);
    }
    int p = 0;
    for (int k0 = 0; k0 < K; k0 += 32, p ^= 1) {
        ewm_wait_async();       // tile k complete (only tile k is outstanding)
        __syncthreads();
        if (k0 + 32 < K) {      // issue tile k+1 into the other buffer
            const char* nA = gAp + (size_t)(k0 + 32) * 2;
            const char* nB = gBp + (size_t)(k0 + 32) * 2;
            char* da = la + (p ^ 1) * LBUF;
            char* db = lb + (p ^ 1) * LBUF;
#pragma unroll
            for (int i = 0; i < 4; ++i) {
                ewm_async_cp16(nA + i * 16, da + i * 16);
                ewm_async_cp16(nB + i * 16, db + i * 16);
            }
        }
#else
    // -------- fallback: register-staged double buffering --------------------
    uint4 ra[4], rb[4];
    {
        const uint4* gA4 = (const uint4*)gAp;
        const uint4* gB4 = (const uint4*)gBp;
#pragma unroll
        for (int i = 0; i < 4; ++i) { ra[i] = gA4[i]; rb[i] = gB4[i]; }
    }
    int p = 0;
    for (int k0 = 0; k0 < K; k0 += 32, p ^= 1) {
        {
            uint4* da = (uint4*)(&lds_a[p][tid * 40]);
            uint4* db = (uint4*)(&lds_b[p][tid * 40]);
#pragma unroll
            for (int i = 0; i < 4; ++i) { da[i] = ra[i]; db[i] = rb[i]; }
        }
        __syncthreads();
        if (k0 + 32 < K) {
            const uint4* nA = (const uint4*)(gAp + (size_t)(k0 + 32) * 2);
            const uint4* nB = (const uint4*)(gBp + (size_t)(k0 + 32) * 2);
#pragma unroll
            for (int i = 0; i < 4; ++i) { ra[i] = nA[i]; rb[i] = nB[i]; }
        }
#endif
        // ---- fragments + 16 WMMAs from buffer p ----
        v16bf afrag[4], bfrag[4];
#pragma unroll
        for (int mi = 0; mi < 4; ++mi) {
            V16BF t;
            const char* q = (const char*)lds_a[p] + ((wrow + mi * 16 + hl) * 40 + kgrpA) * 2;
            t.q[0] = *(const uint4*)q;          // K = kgrpA .. kgrpA+7
            t.q[1] = *(const uint4*)(q + 32);   // K = 16+kgrpA .. 16+kgrpA+7
            afrag[mi] = t.v;
        }
#pragma unroll
        for (int ni = 0; ni < 4; ++ni) {
            V16BF t;
            const char* q = (const char*)lds_b[p] + ((wcol + ni * 16 + hl) * 40 + kgrpB) * 2;
            t.q[0] = *(const uint4*)q;          // K = kgrpB .. kgrpB+7
            t.q[1] = *(const uint4*)(q + 32);   // K = 16+kgrpB .. 16+kgrpB+7
            bfrag[ni] = t.v;
        }
#pragma unroll
        for (int mi = 0; mi < 4; ++mi)
#pragma unroll
            for (int ni = 0; ni < 4; ++ni)
                acc[mi][ni] = __builtin_amdgcn_wmma_f32_16x16x32_bf16(
                    false, afrag[mi], false, bfrag[ni], (short)0, acc[mi][ni], false, false);
    }

    // ---- epilogue: bias add, store f32 z ----
#pragma unroll
    for (int ni = 0; ni < 4; ++ni) {
        int col = n0 + wcol + ni * 16 + hl;
        float bv = biase[col];
#pragma unroll
        for (int mi = 0; mi < 4; ++mi) {
            int rbase = m0 + wrow + mi * 16 + (lane >> 4) * 8;
#pragma unroll
            for (int r = 0; r < 8; ++r)
                Ce[(size_t)(rbase + r) * N + col] = acc[mi][ni][r] + bv;
        }
    }
}

// ---------------------------------------------------------------------------
// LayerNorm + exact GELU (+ optional residual). One block per (e, row).
// ---------------------------------------------------------------------------
__global__ __launch_bounds__(256)
void ewm_ln_gelu(const float* __restrict__ z,
                 const float* __restrict__ g, long long gStrideE,
                 const float* __restrict__ be, long long beStrideE,
                 const float* __restrict__ resid,
                 float* __restrict__ h32, unsigned short* __restrict__ hbf, int Hv) {
    __shared__ float s1[256], s2[256];
    const int e = blockIdx.y, b = blockIdx.x, tid = threadIdx.x;
    const size_t base = ((size_t)e * gridDim.x + b) * (size_t)Hv;
    const float* gp = g  + (size_t)e * gStrideE;
    const float* bp = be + (size_t)e * beStrideE;

    float v[8];
    float sum = 0.f, sq = 0.f;
    const int nPer = Hv / 256;
#pragma unroll
    for (int i = 0; i < 8; ++i) {
        if (i < nPer) {
            float x = z[base + tid + i * 256];
            v[i] = x; sum += x; sq += x * x;
        }
    }
    s1[tid] = sum; s2[tid] = sq;
    __syncthreads();
    for (int off = 128; off; off >>= 1) {
        if (tid < off) { s1[tid] += s1[tid + off]; s2[tid] += s2[tid + off]; }
        __syncthreads();
    }
    const float invH = 1.0f / Hv;
    float mu   = s1[0] * invH;
    float var  = s2[0] * invH - mu * mu;
    float rstd = rsqrtf(var + 1e-5f);
#pragma unroll
    for (int i = 0; i < 8; ++i) {
        if (i < nPer) {
            int idx = tid + i * 256;
            float y  = (v[i] - mu) * rstd * gp[idx] + bp[idx];
            float ge = 0.5f * y * (1.0f + erff(y * 0.70710678118654752f));
            float o  = ge + (resid ? resid[base + idx] : 0.0f);
            h32[base + idx] = o;
            hbf[base + idx] = f2bf_rne(o);
        }
    }
}

// ---------------------------------------------------------------------------
// Reward / done GEMV heads: one 32-lane wave per (e, row).
// ---------------------------------------------------------------------------
__global__ __launch_bounds__(256)
void ewm_rewdone(const float* __restrict__ h,
                 const float* __restrict__ Wr, const float* __restrict__ br,
                 const float* __restrict__ Wd, const float* __restrict__ bd,
                 float* __restrict__ rew, float* __restrict__ don, int Hv, int Bv) {
    const int e    = blockIdx.y;
    const int row  = blockIdx.x * 8 + (threadIdx.x >> 5);
    const int lane = threadIdx.x & 31;
    const float* hp = h  + ((size_t)e * Bv + row) * (size_t)Hv;
    const float* wr = Wr + (size_t)e * Hv;
    const float* wd = Wd + (size_t)e * Hv;
    float ar = 0.f, ad = 0.f;
    for (int k = lane; k < Hv; k += 32) {
        float hv = hp[k];
        ar += hv * wr[k];
        ad += hv * wd[k];
    }
    for (int off = 16; off; off >>= 1) {
        ar += __shfl_down(ar, off);
        ad += __shfl_down(ad, off);
    }
    if (lane == 0) {
        rew[(size_t)e * Bv + row] = ar + br[e];
        don[(size_t)e * Bv + row] = ad + bd[e];
    }
}

// ---------------------------------------------------------------------------
// Cross-ensemble mean + unbiased variance of next_states; uncertainty[b].
// ---------------------------------------------------------------------------
__global__ __launch_bounds__(256)
void ewm_reduce_state(const float* __restrict__ ns, const float* __restrict__ st,
                      float* __restrict__ mean_out, float* __restrict__ unc_out,
                      int Bv, int Sv) {
    __shared__ float red[256];
    const int b = blockIdx.x, tid = threadIdx.x;
    float acc = 0.f;
    for (int s = tid; s < Sv; s += 256) {
        float sv = st[(size_t)b * Sv + s];
        float v[8];
        float sum = 0.f;
#pragma unroll
        for (int e = 0; e < 8; ++e) {
            float x = ns[((size_t)e * Bv + b) * Sv + s] + sv;
            v[e] = x; sum += x;
        }
        float mean = sum * 0.125f;
        mean_out[(size_t)b * Sv + s] = mean;
        float d2 = 0.f;
#pragma unroll
        for (int e = 0; e < 8; ++e) { float d = v[e] - mean; d2 += d * d; }
        acc += d2 * (1.0f / 7.0f);   // ddof = 1
    }
    red[tid] = acc;
    __syncthreads();
    for (int off = 128; off; off >>= 1) {
        if (tid < off) red[tid] += red[tid + off];
        __syncthreads();
    }
    if (tid == 0) unc_out[b] = red[0] / Sv;
}

__global__ __launch_bounds__(256)
void ewm_reduce_rewdone(const float* __restrict__ rew, const float* __restrict__ don,
                        float* __restrict__ mr, float* __restrict__ md, int Bv) {
    int b = blockIdx.x * 256 + threadIdx.x;
    if (b >= Bv) return;
    float sr = 0.f, sd = 0.f;
#pragma unroll
    for (int e = 0; e < 8; ++e) {
        sr += rew[(size_t)e * Bv + b];
        float x = don[(size_t)e * Bv + b];
        sd += 1.0f / (1.0f + expf(-x));
    }
    mr[b] = sr * 0.125f;
    md[b] = sd * 0.125f;
}

// ---------------------------------------------------------------------------
extern "C" void kernel_launch(void* const* d_in, const int* in_sizes, int n_in,
                              void* d_out, int out_size, void* d_ws, size_t ws_size,
                              hipStream_t stream) {
    (void)in_sizes; (void)n_in; (void)out_size; (void)ws_size;

    constexpr int B = 8192, S = 512, Adim = 64, H = 2048, E = 8, L = 3;
    constexpr int SA = S + Adim;   // 576

    const float* state   = (const float*)d_in[0];
    const float* action  = (const float*)d_in[1];
    const float* W_in    = (const float*)d_in[2];
    const float* b_in    = (const float*)d_in[3];
    const float* g_in    = (const float*)d_in[4];
    const float* be_in   = (const float*)d_in[5];
    const float* W_h     = (const float*)d_in[6];
    const float* b_h     = (const float*)d_in[7];
    const float* g_h     = (const float*)d_in[8];
    const float* be_h    = (const float*)d_in[9];
    const float* W_state = (const float*)d_in[10];
    const float* b_state = (const float*)d_in[11];
    const float* W_rew   = (const float*)d_in[12];
    const float* b_rew   = (const float*)d_in[13];
    const float* W_done  = (const float*)d_in[14];
    const float* b_done  = (const float*)d_in[15];
    float* out = (float*)d_out;

    // workspace carve-up (256B aligned)
    char* ws = (char*)d_ws;
    size_t off = 0;
    auto carve = [&](size_t bytes) -> char* {
        char* p = ws + off;
        off = (off + bytes + 255) & ~(size_t)255;
        return p;
    };
    unsigned short* xcat  = (unsigned short*)carve((size_t)B * SA * 2);
    float*          zbuf  = (float*)carve((size_t)E * B * H * 4);
    float*          h32   = (float*)carve((size_t)E * B * H * 4);
    unsigned short* hbf   = (unsigned short*)carve((size_t)E * B * H * 2);
    float*          nsb   = (float*)carve((size_t)E * B * S * 4);
    float*          rwb   = (float*)carve((size_t)E * B * 4);
    float*          dnb   = (float*)carve((size_t)E * B * 4);
    unsigned short* wtIn  = (unsigned short*)carve((size_t)E * SA * H * 2);      // [E][H][SA]
    unsigned short* wtH   = (unsigned short*)carve((size_t)E * L * H * H * 2);   // [E][L][H][H]
    unsigned short* wtSt  = (unsigned short*)carve((size_t)E * H * S * 2);       // [E][S][H]

    // 0) one-shot weight transpose+convert f32 [K][N] -> bf16 [N][K]
    ewm_wtrans_bf16<<<dim3(H / 32, SA / 32, E), 256, 0, stream>>>(W_in, wtIn, SA, H);
    ewm_wtrans_bf16<<<dim3(H / 32, H / 32, E * L), 256, 0, stream>>>(W_h, wtH, H, H);
    ewm_wtrans_bf16<<<dim3(S / 32, H / 32, E), 256, 0, stream>>>(W_state, wtSt, H, S);

    // 1) pack concat(state, action) -> bf16
    ewm_pack_x<<<(B * SA + 255) / 256, 256, 0, stream>>>(state, action, xcat, B, S, Adim);

    // 2) input projection: z[e] = x @ W_in[e] + b_in[e]
    dim3 gIn(H / 128, B / 128, E);
    ewm_gemm_bias<<<gIn, 128, 0, stream>>>(
        xcat, 0LL, wtIn, (long long)SA * H, b_in, (long long)H,
        zbuf, (long long)B * H, B, H, SA);
    ewm_ln_gelu<<<dim3(B, E), 256, 0, stream>>>(
        zbuf, g_in, (long long)H, be_in, (long long)H, nullptr, h32, hbf, H);

    // 3) residual hidden blocks
    for (int l = 0; l < L; ++l) {
        ewm_gemm_bias<<<gIn, 128, 0, stream>>>(
            hbf, (long long)B * H,
            wtH + (size_t)l * H * H, (long long)L * H * H,
            b_h + (size_t)l * H, (long long)L * H,
            zbuf, (long long)B * H, B, H, H);
        ewm_ln_gelu<<<dim3(B, E), 256, 0, stream>>>(
            zbuf, g_h + (size_t)l * H, (long long)L * H,
            be_h + (size_t)l * H, (long long)L * H,
            h32, h32, hbf, H);
    }

    // 4) state head: ns[e] = h[e] @ W_state[e] + b_state[e]
    dim3 gSt(S / 128, B / 128, E);
    ewm_gemm_bias<<<gSt, 128, 0, stream>>>(
        hbf, (long long)B * H, wtSt, (long long)H * S, b_state, (long long)S,
        nsb, (long long)B * S, B, S, H);

    // 5) reward / done GEMV heads
    ewm_rewdone<<<dim3(B / 8, E), 256, 0, stream>>>(
        h32, W_rew, b_rew, W_done, b_done, rwb, dnb, H, B);

    // 6) reductions into d_out: [mean_state | mean_reward | mean_done | uncertainty]
    float* out_ms  = out;
    float* out_mr  = out + (size_t)B * S;
    float* out_md  = out + (size_t)B * S + B;
    float* out_unc = out + (size_t)B * S + 2 * (size_t)B;
    ewm_reduce_state<<<B, 256, 0, stream>>>(nsb, state, out_ms, out_unc, B, S);
    ewm_reduce_rewdone<<<B / 256, 256, 0, stream>>>(rwb, dnb, out_mr, out_md, B);
}
